// CRF_29343216566876
// MI455X (gfx1250) — matrix operand order repaired and checked by reading, not compiled, not run on hardware
//
#include <hip/hip_runtime.h>

typedef _Float16 v16h __attribute__((ext_vector_type(16)));
typedef _Float16 v8h  __attribute__((ext_vector_type(8)));
typedef float    v8f  __attribute__((ext_vector_type(8)));

#define NPATCH 4096
#define BATCH  4
#define NCOL   16   // 12 used (4 batches x 3 vectors), 4 zero-padded
#define ITERS  10

// ---------------------------------------------------------------------------
// Kernel A (once): W_sym = 0.5*(W + W^T), stored f16.  W (64MB) is L2-resident
// so the transposed read mostly hits L2.
// ---------------------------------------------------------------------------
__global__ void sym_kernel(const float* __restrict__ W, _Float16* __restrict__ Wsym) {
    const int idx = blockIdx.x * 256 + threadIdx.x;     // 0 .. 2^24-1
    const int i = idx >> 12;
    const int j = idx & (NPATCH - 1);
    const float v = 0.5f * (W[idx] + W[j * NPATCH + i]);
    Wsym[idx] = (_Float16)v;
}

// ---------------------------------------------------------------------------
// Kernel B (per iter): build transposed activation matrix Xt[col][k], f16.
// col = b*3 + {0: g*m, 1: g*m*ux, 2: g*m*uy}; cols 12..15 zero.
// m = 2*sigmoid(l) - 1 = tanh(l/2).
// ---------------------------------------------------------------------------
__global__ void buildx_kernel(const float* __restrict__ dp,
                              const float* __restrict__ l,
                              _Float16* __restrict__ Xt) {
    const int t = blockIdx.x * 256 + threadIdx.x;       // 0 .. 16383
    const int b = t >> 12;
    const int n = t & (NPATCH - 1);
    const float fx = dp[2 * t + 0];
    const float fy = dp[2 * t + 1];
    const float sq = fx * fx + fy * fy;
    const float inv = rsqrtf(sq);
    const float g = expf(-0.5f * sq);
    const float m = tanhf(0.5f * l[t]);
    const float gm = g * m;
    Xt[(size_t)(b * 3 + 0) * NPATCH + n] = (_Float16)gm;
    Xt[(size_t)(b * 3 + 1) * NPATCH + n] = (_Float16)(gm * fx * inv);
    Xt[(size_t)(b * 3 + 2) * NPATCH + n] = (_Float16)(gm * fy * inv);
    if (b == 0) {
        Xt[(size_t)12 * NPATCH + n] = (_Float16)0.f;
        Xt[(size_t)13 * NPATCH + n] = (_Float16)0.f;
        Xt[(size_t)14 * NPATCH + n] = (_Float16)0.f;
        Xt[(size_t)15 * NPATCH + n] = (_Float16)0.f;
    }
}

// ---------------------------------------------------------------------------
// Kernel C (per iter): Y[4096x16] = Wsym[4096x4096] @ X[4096x16] via
// v_wmma_f32_16x16x32_f16.  One block per 16-row tile; 8 waves split K,
// partial 16x16 f32 tiles reduced through LDS.
// ---------------------------------------------------------------------------
__global__ void __launch_bounds__(256) gemm_kernel(const _Float16* __restrict__ A,
                                                   const _Float16* __restrict__ Xt,
                                                   float* __restrict__ Y) {
    __shared__ float lds[8 * 256];

    const int tid   = threadIdx.x;
    const int wave  = tid >> 5;
    const int lane  = tid & 31;
    const int row16 = lane & 15;
    const int khalf = lane >> 4;          // 0 or 1
    const int tile  = blockIdx.x;         // 0..255

    const size_t arow = (size_t)(tile * 16 + row16) * NPATCH;  // A row for this lane
    const size_t bcol = (size_t)row16 * NPATCH;                // Xt row = output column

    v8f c = {};
    const int k0 = wave * 512;

    #pragma unroll 4
    for (int k = k0; k < k0 + 512; k += 32) {
        // A layout (16x32 f16): lanes 0-15 hold K0..15, lanes 16-31 hold K16..31
        // (contiguous along K, matching B).  Two 16B chunks per lane.
        v8h alo = *(const v8h*)(A + arow + k + 16 * khalf);
        v8h ahi = *(const v8h*)(A + arow + k + 16 * khalf + 8);
        v16h a = __builtin_shufflevector(alo, ahi,
                 0,1,2,3,4,5,6,7,8,9,10,11,12,13,14,15);
        // B layout (32x16 f16): lanes 0-15 hold K0..15, lanes 16-31 hold K16..31,
        // contiguous along K with Xt stored [col][k] -> one 32B per-lane chunk.
        v16h b = *(const v16h*)(Xt + bcol + k + 16 * khalf);

        c = __builtin_amdgcn_wmma_f32_16x16x32_f16(
                /*neg_a=*/false, a, /*neg_b=*/false, b,
                /*c_mod=*/(short)0, c, /*reuse_a=*/false, /*reuse_b=*/false);
    }

    #pragma unroll
    for (int v = 0; v < 8; ++v)
        lds[wave * 256 + v * 32 + lane] = c[v];
    __syncthreads();

    float s = 0.f;
    #pragma unroll
    for (int w = 0; w < 8; ++w)
        s += lds[w * 256 + tid];

    // (vgpr,lane) -> (M,N): M = vgpr + 8*(lane>=16), N = lane&15
    const int v  = tid >> 5;
    const int M  = v + 8 * ((tid >> 4) & 1);
    const int nc = tid & 15;
    Y[(size_t)(tile * 16 + M) * NCOL + nc] = s;
}

// ---------------------------------------------------------------------------
// Kernel D (per iter): l_out = unary + g*(y0 - ux*y1 - uy*y2)
// ---------------------------------------------------------------------------
__global__ void update_kernel(const float* __restrict__ unary,
                              const float* __restrict__ dp,
                              const float* __restrict__ Y,
                              float* __restrict__ out) {
    const int t = blockIdx.x * 256 + threadIdx.x;
    const int b = t >> 12;
    const int n = t & (NPATCH - 1);
    const float fx = dp[2 * t + 0];
    const float fy = dp[2 * t + 1];
    const float sq = fx * fx + fy * fy;
    const float inv = rsqrtf(sq);
    const float g = expf(-0.5f * sq);
    const float ux = fx * inv;
    const float uy = fy * inv;
    const float* y = Y + (size_t)n * NCOL + b * 3;
    const float E = g * (y[0] - ux * y[1] - uy * y[2]);
    out[t] = unary[t] + E;
}

extern "C" void kernel_launch(void* const* d_in, const int* in_sizes, int n_in,
                              void* d_out, int out_size, void* d_ws, size_t ws_size,
                              hipStream_t stream) {
    const float* dp     = (const float*)d_in[0];   // (4,64,64,2)
    const float* logits = (const float*)d_in[1];   // (4,4096,1)
    const float* W      = (const float*)d_in[2];   // (1,4096,4096)
    float*       out    = (float*)d_out;           // (4,4096,1)

    char* ws = (char*)d_ws;
    _Float16* Wsym = (_Float16*)ws;                                       // 32 MB
    _Float16* Xt   = (_Float16*)(ws + (size_t)NPATCH * NPATCH * 2);       // 128 KB
    float*    Y    = (float*)((char*)Xt + (size_t)NCOL * NPATCH * 2);     // 256 KB
    float*    lcur = (float*)((char*)Y  + (size_t)NPATCH * NCOL * 4);     // 64 KB

    sym_kernel<<<(NPATCH * NPATCH) / 256, 256, 0, stream>>>(W, Wsym);

    for (int it = 0; it < ITERS; ++it) {
        const float* lsrc = (it == 0) ? logits : lcur;
        float*       ldst = (it == ITERS - 1) ? out : lcur;
        buildx_kernel<<<(BATCH * NPATCH) / 256, 256, 0, stream>>>(dp, lsrc, Xt);
        gemm_kernel<<<NPATCH / 16, 256, 0, stream>>>(Wsym, Xt, Y);
        update_kernel<<<(BATCH * NPATCH) / 256, 256, 0, stream>>>(logits, dp, Y, ldst);
    }
}
